// EmbedMatcher_27393301414159
// MI455X (gfx1250) — matrix operand order repaired and checked by reading, not compiled, not run on hardware
//
#include <hip/hip_runtime.h>
#include <stdint.h>

typedef __attribute__((ext_vector_type(16))) __bf16 v16bf;
typedef __attribute__((ext_vector_type(8)))  float  v8f;

#define D      128
#define KX     384      // [q(128) | hq(128) | r(128)]
#define FEWN   5
#define NEIGH  200
#define STEPS  4

union Frag { v16bf v; uint4 u[2]; };

__device__ __forceinline__ unsigned short f2bf(float f){
  union { float f; uint32_t u; } c; c.f = f;
  uint32_t u = c.u;
  u += 0x7FFFu + ((u >> 16) & 1u);     // round-to-nearest-even
  return (unsigned short)(u >> 16);
}
// single v_rcp_f32 instead of the IEEE div expansion
__device__ __forceinline__ float rcpf(float x){ return __builtin_amdgcn_rcpf(x); }
__device__ __forceinline__ float sigf(float x){
  return rcpf(1.0f + __expf(-x));
}
// branch-free tanh on v_exp_f32 + v_rcp_f32: exact limits at +/-inf
__device__ __forceinline__ float tanhfast(float x){
  return 1.0f - 2.0f * rcpf(__expf(2.0f * x) + 1.0f);
}
__device__ __forceinline__ v8f splat8(float x){
  v8f v;
  #pragma unroll
  for (int i = 0; i < 8; ++i) v[i] = x;
  return v;
}

// ---------------------------------------------------------------------------
// Prep: W_cat = bf16([W_ih | W_hh]) row-major (1024 x 384); bias = b_ih + b_hh
// ---------------------------------------------------------------------------
__global__ __launch_bounds__(128) void prep_weights(
    const float* __restrict__ W_ih, const float* __restrict__ W_hh,
    const float* __restrict__ b_ih, const float* __restrict__ b_hh,
    unsigned short* __restrict__ Wcat, float* __restrict__ bias){
  const int row = blockIdx.x;        // 0..1023 (gate unit)
  const int t   = threadIdx.x;       // 0..127
  Wcat[row*KX + t]       = f2bf(W_ih[row*128 + t]);
  Wcat[row*KX + 128 + t] = f2bf(W_hh[row*256 + t]);
  Wcat[row*KX + 256 + t] = f2bf(W_hh[row*256 + 128 + t]);
  if (t == 0) bias[row] = b_ih[row] + b_hh[row];
}

// ---------------------------------------------------------------------------
// Support branch: GCN-sum (linearity: sum neighbors first) -> tanh -> MLP ->
// LayerNorm -> support_g (5x128) and mean_support (128). One workgroup.
// ---------------------------------------------------------------------------
__global__ __launch_bounds__(128) void support_branch(
    const int* __restrict__ sp, const float* __restrict__ emb,
    const float* __restrict__ gcnW, const float* __restrict__ gcnb,
    const float* __restrict__ p1W,  const float* __restrict__ p1b,
    const float* __restrict__ p2W,  const float* __restrict__ p2b,
    const float* __restrict__ lng,  const float* __restrict__ lnb,
    float* __restrict__ supg, float* __restrict__ ms){
  __shared__ float csum[FEWN][256];
  __shared__ float sup [FEWN][128];
  __shared__ float h1  [FEWN][256];
  __shared__ float xb  [FEWN][128];
  __shared__ float stats[FEWN][2];
  const int t = threadIdx.x;

  for (int f = 0; f < FEWN; ++f){
    float c0 = 0.f, c1 = 0.f;
    for (int j = 0; j < NEIGH; ++j){
      const long i0 = sp[(f*NEIGH + j)*2 + 0];
      const long i1 = sp[(f*NEIGH + j)*2 + 1];
      c0 += emb[i0*D + t];
      c1 += emb[i1*D + t];
    }
    csum[f][t] = c0;
    csum[f][128 + t] = c1;
  }
  __syncthreads();
  for (int f = 0; f < FEWN; ++f){               // msg.sum = csum @ Wᵀ + 200*b
    float a = 0.f;
    for (int k = 0; k < 256; ++k) a += csum[f][k] * gcnW[t*256 + k];
    sup[f][t] = tanhf((a + 200.0f * gcnb[t]) * 0.2f);
  }
  __syncthreads();
  for (int jj = 0; jj < 2; ++jj){
    const int j = t + jj*128;
    for (int f = 0; f < FEWN; ++f){
      float a = 0.f;
      for (int k = 0; k < D; ++k) a += sup[f][k] * p1W[j*D + k];
      a += p1b[j];
      h1[f][j] = a > 0.f ? a : 0.f;
    }
  }
  __syncthreads();
  for (int f = 0; f < FEWN; ++f){
    float a = 0.f;
    for (int k = 0; k < 256; ++k) a += h1[f][k] * p2W[t*256 + k];
    xb[f][t] = a + p2b[t] + sup[f][t];
  }
  __syncthreads();
  if (t < FEWN){
    float mu = 0.f;
    for (int k = 0; k < D; ++k) mu += xb[t][k];
    mu *= (1.0f/D);
    float var = 0.f;
    for (int k = 0; k < D; ++k){ const float d = xb[t][k] - mu; var += d*d; }
    stats[t][0] = mu;
    stats[t][1] = rsqrtf(var * (1.0f/D) + 1e-5f);
  }
  __syncthreads();
  float acc = 0.f;
  for (int f = 0; f < FEWN; ++f){
    const float g = lng[t] * (xb[f][t] - stats[f][0]) * stats[f][1] + lnb[t];
    supg[f*D + t] = g;
    acc += g;
  }
  ms[t] = acc * (1.0f/FEWN);
}

// ---------------------------------------------------------------------------
// Main: 16 query rows per block, 4 waves, 4 fused recurrent steps.
// gates(16x1024) = X(16x384,bf16) @ W_catᵀ via v_wmma_f32_16x16x32_bf16.
// Wave w owns hidden columns [w*64, w*64+64); computes i/f/g/o tiles there.
// kt loop deliberately NOT unrolled: keeps <=1 A-frag + 4 B-frags live so the
// kernel stays inside 256 VGPRs (no spills / no vgpr-msb mode).
// ---------------------------------------------------------------------------
__global__ __launch_bounds__(128) void matcher_main(
    const int* __restrict__ qp, const float* __restrict__ emb,
    const float* __restrict__ supg, const float* __restrict__ ms,
    const float* __restrict__ bias, const unsigned short* __restrict__ Wcat,
    float* __restrict__ out){
  __shared__ __align__(16) unsigned short Xs[16*KX];  // bf16 [row][k]
  __shared__ float qs [16*D];
  __shared__ float hqs[16*D];
  __shared__ float sgs[FEWN*D];
  __shared__ float dots[16];

  const int t    = threadIdx.x;
  const int wave = t >> 5;
  const int lane = t & 31;
  const int r0   = blockIdx.x * 16;

  for (int i = t; i < FEWN*D; i += 128) sgs[i] = supg[i];

  { // gather q rows, init X = [q, 0, 0]
    const int row = t >> 3;
    const int c0  = (t & 7) * 16;
    const long eidx = (long)qp[r0 + row] * D;
    #pragma unroll
    for (int e = 0; e < 16; ++e){
      const int col = c0 + e;
      const float v = emb[eidx + col];
      qs[row*D + col]        = v;
      Xs[row*KX + col]       = f2bf(v);
      Xs[row*KX + 128 + col] = 0;
      Xs[row*KX + 256 + col] = 0;
    }
  }
  __syncthreads();

  const int colw  = wave * 64;
  const int nsel  = lane & 15;    // N (and M for A-frag)
  const int khalf = lane >> 4;    // lane-half selects K sub-chunk

  // step-invariant: bias scalars and per-gate weight row base pointers
  float biasr[4][4];
  #pragma unroll
  for (int nl = 0; nl < 4; ++nl)
    #pragma unroll
    for (int G = 0; G < 4; ++G)
      biasr[nl][G] = bias[G*256 + colw + nl*16 + nsel];

  v8f cst[4];
  #pragma unroll
  for (int nl = 0; nl < 4; ++nl) cst[nl] = splat8(0.0f);

  for (int s = 0; s < STEPS; ++s){
    #pragma unroll
    for (int nl = 0; nl < 4; ++nl){
      const int colbase = colw + nl*16;
      const unsigned short* wp[4];
      #pragma unroll
      for (int G = 0; G < 4; ++G)
        wp[G] = Wcat + (size_t)(G*256 + colbase + nsel)*KX + khalf*16;

      v8f acc[4];
      #pragma unroll
      for (int G = 0; G < 4; ++G) acc[G] = splat8(biasr[nl][G]);

      const unsigned short* xrow = Xs + nsel*KX + khalf*16;
      #pragma unroll 1
      for (int kt = 0; kt < 12; ++kt){
        const int kb = kt * 32;
        if (kt + 2 < 12)   // keep the weight stream warm
          __builtin_prefetch(wp[0] + kb + 64, 0, 1);
        Frag a;            // A: lane m=nsel, contiguous 16 K at kb + khalf*16
        a.u[0] = *reinterpret_cast<const uint4*>(xrow + kb);
        a.u[1] = *reinterpret_cast<const uint4*>(xrow + kb + 8);
        #pragma unroll
        for (int G = 0; G < 4; ++G){
          Frag b;          // B: lane n, contiguous 16 K at kb + khalf*16
          const uint4* pb = reinterpret_cast<const uint4*>(wp[G] + kb);
          b.u[0] = pb[0];
          b.u[1] = pb[1];
          acc[G] = __builtin_amdgcn_wmma_f32_16x16x32_bf16(
              false, a.v, false, b.v, (short)0, acc[G], false, false);
        }
      }
      // LSTM pointwise in accumulator layout: M = e + khalf*8, N = nsel
      const bool isHq = (colbase < 128);   // wave-uniform
      #pragma unroll
      for (int e = 0; e < 8; ++e){
        const float iv = sigf(acc[0][e]);
        const float fv = sigf(acc[1][e]);
        const float gv = tanhfast(acc[2][e]);
        const float c  = fv * cst[nl][e] + iv * gv;
        cst[nl][e] = c;
        if (isHq){
          const float h   = sigf(acc[3][e]) * tanhfast(c);
          const int colN  = colbase + nsel;
          const int row   = e + khalf*8;
          hqs[row*D + colN] = qs[row*D + colN] + h;
        }
      }
    }
    __syncthreads();

    if (s < STEPS - 1){
      // publish hq as bf16 into X (safe: no one reads X between barriers)
      for (int i = t; i < 16*D; i += 128){
        const int row = i >> 7, col = i & 127;
        Xs[row*KX + 128 + col] = f2bf(hqs[i]);
      }
      // attention over the 5 support rows; r -> X[:,256:384]
      if (t < 16){
        float lg[FEWN];
        float mx = -3.0e38f;
        for (int f = 0; f < FEWN; ++f){
          float a = 0.f;
          for (int k = 0; k < D; ++k) a += hqs[t*D + k] * sgs[f*D + k];
          lg[f] = a; if (a > mx) mx = a;
        }
        float se = 0.f;
        for (int f = 0; f < FEWN; ++f){ lg[f] = __expf(lg[f] - mx); se += lg[f]; }
        const float inv = rcpf(se);
        for (int k = 0; k < D; ++k){
          float r = 0.f;
          for (int f = 0; f < FEWN; ++f) r += lg[f] * sgs[f*D + k];
          Xs[t*KX + 256 + k] = f2bf(r * inv);
        }
      }
      __syncthreads();
    }
  }

  // out[b] = mean(hq[2b], hq[2b+1]) . mean_support
  if (t < 16){
    float a = 0.f;
    for (int k = 0; k < D; ++k) a += hqs[t*D + k] * ms[k];
    dots[t] = a;
  }
  __syncthreads();
  if (t < 8) out[blockIdx.x*8 + t] = 0.5f * (dots[2*t] + dots[2*t + 1]);
}

// ---------------------------------------------------------------------------
extern "C" void kernel_launch(void* const* d_in, const int* in_sizes, int n_in,
                              void* d_out, int out_size, void* d_ws, size_t ws_size,
                              hipStream_t stream){
  const int*   qp   = (const int*)  d_in[0];
  const int*   sp   = (const int*)  d_in[1];
  const float* emb  = (const float*)d_in[2];
  const float* gcnW = (const float*)d_in[3];
  const float* gcnb = (const float*)d_in[4];
  const float* p1W  = (const float*)d_in[5];
  const float* p1b  = (const float*)d_in[6];
  const float* p2W  = (const float*)d_in[7];
  const float* p2b  = (const float*)d_in[8];
  const float* lng  = (const float*)d_in[9];
  const float* lnb  = (const float*)d_in[10];
  const float* W_ih = (const float*)d_in[11];
  const float* W_hh = (const float*)d_in[12];
  const float* b_ih = (const float*)d_in[13];
  const float* b_hh = (const float*)d_in[14];
  float* out = (float*)d_out;

  char* ws = (char*)d_ws;
  float*          supgW = (float*)(ws);            // 5*128 f32   = 2560 B
  float*          msW   = (float*)(ws + 2560);     // 128 f32     = 512 B
  float*          biasW = (float*)(ws + 3072);     // 1024 f32    = 4096 B
  unsigned short* WcatW = (unsigned short*)(ws + 7168); // 1024*384 bf16 = 768 KB

  prep_weights  <<<1024, 128, 0, stream>>>(W_ih, W_hh, b_ih, b_hh, WcatW, biasW);
  support_branch<<<   1, 128, 0, stream>>>(sp, emb, gcnW, gcnb, p1W, p1b,
                                           p2W, p2b, lng, lnb, supgW, msW);
  matcher_main  <<<2048, 128, 0, stream>>>(qp, emb, supgW, msW, biasW, WcatW, out);
  (void)in_sizes; (void)n_in; (void)out_size; (void)ws_size;
}
